// HarsanyiNet_5952824672511
// MI455X (gfx1250) — matrix operand order repaired and checked by reading, not compiled, not run on hardware
//
#include <hip/hip_runtime.h>
#include <hip/hip_bf16.h>

// ---------------------------------------------------------------------------
// HarsanyiNet forward on MI455X (gfx1250, wave32).
//
// prod_i de[b,o,i] = exp( sum_i M[o,i]*log(tanh(G|x[b,i]|)) )  with M=(v>0)
// => per layer, two GEMMs sharing one streaming pass over the f32 weights:
//      s = L @ M^T ,  h = X @ (fc.*M)^T ,  out = relu(h*exp(s))
// Roofline: 16 MB of weights read once / 23.3 TB/s ~ 0.7 us; 0.54 GFLOP is
// noise vs the bf16 WMMA ceiling -> purely HBM-bound.  Hence:
//  - weights f32->bf16 converted in-register (no extra prep-pass traffic),
//  - register double-buffering of weight tiles (loads overlap WMMA),
//  - activations async-staged to LDS once per block (ASYNCcnt path,
//    4-way reuse across waves, inner loop reads via ds_load_b128),
//  - split-K=4 -> 64 blocks x 4 waves to keep HBM saturated.
// ---------------------------------------------------------------------------

#define BB    64
#define K_DIM 1024           // IN == HID == 1024 for both layers
#define O_DIM 1024
#define NC    10
#define GAMMA 100.0f
#define KSPLIT 4
#define KLEN  (K_DIM / KSPLIT)          // 256
#define LDSS  (KLEN + 8)                // padded LDS row stride (ushorts)

#ifndef __has_builtin
#define __has_builtin(x) 0
#endif
#if __has_builtin(__builtin_amdgcn_global_load_async_to_lds_b128)
#define HAVE_ASYNC_LDS 1
#else
#define HAVE_ASYNC_LDS 0
#endif

typedef __attribute__((ext_vector_type(16))) __bf16 v16bf;
typedef __attribute__((ext_vector_type(8)))  float  v8f;
typedef __attribute__((ext_vector_type(4)))  int    v4i;

union FragBF {            // one WMMA bf16 operand: 16 bf16 = 8 VGPRs
    v16bf    v;
    uint4    q[2];
    unsigned u[8];
};

__device__ __forceinline__ unsigned short f2bf(float f) {
    unsigned u = __builtin_bit_cast(unsigned, f);
    unsigned r = u + 0x7FFFu + ((u >> 16) & 1u);      // RNE f32->bf16
    return (unsigned short)(r >> 16);
}

__device__ __forceinline__ unsigned pack_bf2(float lo, float hi) {
#if __has_builtin(__builtin_amdgcn_cvt_pk_bf16_f32)
    typedef __attribute__((ext_vector_type(2))) __bf16 v2bf;
    v2bf p = __builtin_amdgcn_cvt_pk_bf16_f32(lo, hi);
    return __builtin_bit_cast(unsigned, p);
#else
    return (unsigned)f2bf(lo) | ((unsigned)f2bf(hi) << 16);
#endif
}

__device__ __forceinline__ void act_to_bf(float a, unsigned short* Xb,
                                          unsigned short* Lb, int i) {
    float t = tanhf(GAMMA * fabsf(a));
    float l = fmaxf(__logf(fmaxf(t, 1e-30f)), -60.0f);
    Xb[i] = f2bf(a);
    Lb[i] = f2bf(l);
}

// ---- prep: x -> bf16 activations + bf16 clamped log-tanh ------------------
__global__ void prep_input(const float* __restrict__ x,
                           unsigned short* __restrict__ Xb,
                           unsigned short* __restrict__ Lb, int n) {
    int i = blockIdx.x * blockDim.x + threadIdx.x;
    if (i >= n) return;
    act_to_bf(x[i], Xb, Lb, i);
}

// ---- helpers for the fused dual-GEMM --------------------------------------
__device__ __forceinline__ void stage_lds(unsigned short* lds,
                                          const unsigned short* g,
                                          int kBeg, int tid) {
    // 64 rows x KLEN cols in 16-byte chunks; 128 threads cooperate.
    for (int c = tid; c < 64 * (KLEN / 8); c += 128) {
        const int row = c >> 5, cc = c & 31;                  // KLEN/8 == 32
        const unsigned short* src = g + (size_t)row * K_DIM + kBeg + cc * 8;
        unsigned short*       dst = lds + row * LDSS + cc * 8;
#if HAVE_ASYNC_LDS
        unsigned short* s = const_cast<unsigned short*>(src);
        __builtin_amdgcn_global_load_async_to_lds_b128(
            (__attribute__((address_space(1))) v4i*)s,
            (__attribute__((address_space(3))) v4i*)dst, 0, 0);
#else
        *(uint4*)dst = *(const uint4*)src;
#endif
    }
}

__device__ __forceinline__ void wait_stage() {
#if HAVE_ASYNC_LDS
#if __has_builtin(__builtin_amdgcn_s_wait_asynccnt)
    __builtin_amdgcn_s_wait_asynccnt(0);
#else
    asm volatile("s_wait_asynccnt 0x0" ::: "memory");
#endif
#endif
    __syncthreads();
}

__device__ __forceinline__ void load_w(const float* vrow, const float* frow,
                                       int kb, float4 vv[4], float4 ff[4]) {
#pragma unroll
    for (int c = 0; c < 4; ++c) {
        vv[c] = *(const float4*)(vrow + kb + 4 * c);
        ff[c] = *(const float4*)(frow + kb + 4 * c);
    }
}

__device__ __forceinline__ void kstep(const unsigned short* Xs,
                                      const unsigned short* Ls,
                                      int kk, int lr, int lg,
                                      const float4 vv[4], const float4 ff[4],
                                      v8f acc_h[4], v8f acc_s[4]) {
    // Build B fragments (mask 0/1, fc.*mask) from the prefetched f32 tiles.
    FragBF bm, bw;
#pragma unroll
    for (int c = 0; c < 4; ++c) {
        const float m0 = vv[c].x > 0.f ? 1.f : 0.f;
        const float m1 = vv[c].y > 0.f ? 1.f : 0.f;
        const float m2 = vv[c].z > 0.f ? 1.f : 0.f;
        const float m3 = vv[c].w > 0.f ? 1.f : 0.f;
        bm.u[2 * c]     = pack_bf2(m0, m1);
        bm.u[2 * c + 1] = pack_bf2(m2, m3);
        bw.u[2 * c]     = pack_bf2(ff[c].x * m0, ff[c].y * m1);
        bw.u[2 * c + 1] = pack_bf2(ff[c].z * m2, ff[c].w * m3);
    }
#pragma unroll
    for (int mt = 0; mt < 4; ++mt) {
        const unsigned short* xr = Xs + (mt * 16 + lr) * LDSS;
        const unsigned short* lw = Ls + (mt * 16 + lr) * LDSS;
        FragBF ax, al;                                  // A frags: contiguous k = 16*lg + s
        ax.q[0] = *(const uint4*)(xr + kk +     16 * lg);
        ax.q[1] = *(const uint4*)(xr + kk + 8 + 16 * lg);
        al.q[0] = *(const uint4*)(lw + kk +     16 * lg);
        al.q[1] = *(const uint4*)(lw + kk + 8 + 16 * lg);
        acc_h[mt] = __builtin_amdgcn_wmma_f32_16x16x32_bf16(
            false, ax.v, false, bw.v, (short)0, acc_h[mt], false, false);
        acc_s[mt] = __builtin_amdgcn_wmma_f32_16x16x32_bf16(
            false, al.v, false, bm.v, (short)0, acc_s[mt], false, false);
    }
}

// ---- fused dual-GEMM over one K-slice -------------------------------------
__global__ void __launch_bounds__(128)
harsanyi_gemm(const unsigned short* __restrict__ Xb,  // [B,K] bf16
              const unsigned short* __restrict__ Lb,  // [B,K] bf16 log-tanh
              const float* __restrict__ Vw,           // [O,K] f32
              const float* __restrict__ Fw,           // [O,K] f32
              float* __restrict__ part_h,             // [KSPLIT,B,O]
              float* __restrict__ part_s)
{
    __shared__ unsigned short Xs[64 * LDSS];
    __shared__ unsigned short Ls[64 * LDSS];

    const int tid  = threadIdx.x;
    const int wave = tid >> 5;
    const int lane = tid & 31;
    const int lr   = lane & 15;
    const int lg   = lane >> 4;
    const int n0   = (blockIdx.x * 4 + wave) * 16;   // output column base
    const int kBeg = blockIdx.y * KLEN;

    // Async-stage this K-slice of activations into LDS (shared by 4 waves).
    stage_lds(Xs, Xb, kBeg, tid);
    stage_lds(Ls, Lb, kBeg, tid);
    wait_stage();

    const float* vrow = Vw + (size_t)(n0 + lr) * K_DIM;
    const float* frow = Fw + (size_t)(n0 + lr) * K_DIM;

    v8f acc_h[4], acc_s[4];
#pragma unroll
    for (int mt = 0; mt < 4; ++mt) { acc_h[mt] = {}; acc_s[mt] = {}; }

    // Software-pipelined: weight tiles double-buffered in registers so the
    // next step's global_load_b128s overlap the current step's WMMAs.
    float4 va[4], fa[4], vb[4], fb[4];
    load_w(vrow, frow, kBeg + 16 * lg, va, fa);
#pragma unroll 2
    for (int kk = 0; kk < KLEN; kk += 64) {
        const int kb = kBeg + kk + 16 * lg;
        load_w(vrow, frow, kb + 32, vb, fb);
        kstep(Xs, Ls, kk, lr, lg, va, fa, acc_h, acc_s);
        if (kk + 64 < KLEN) load_w(vrow, frow, kb + 64, va, fa);
        kstep(Xs, Ls, kk + 32, lr, lg, vb, fb, acc_h, acc_s);
    }

    // C/D layout: lane -> col n0+lr; VGPR r -> row mt*16 + r + 8*lg.
    const size_t base = (size_t)blockIdx.y * ((size_t)BB * O_DIM);
#pragma unroll
    for (int mt = 0; mt < 4; ++mt) {
#pragma unroll
        for (int r = 0; r < 8; ++r) {
            const size_t idx = base + (size_t)(mt * 16 + r + 8 * lg) * O_DIM
                             + n0 + lr;
            part_h[idx] = acc_h[mt][r];
            part_s[idx] = acc_s[mt][r];
        }
    }
}

// ---- K-split reduce + activation epilogue (+ next-layer bf16 prep) --------
__global__ void combine(const float* __restrict__ part_h,
                        const float* __restrict__ part_s,
                        float* __restrict__ hout,
                        unsigned short* __restrict__ Hb,   // may be null
                        unsigned short* __restrict__ Lb,   // may be null
                        int n)
{
    int i = blockIdx.x * blockDim.x + threadIdx.x;
    if (i >= n) return;
    float h = 0.f, s = 0.f;
#pragma unroll
    for (int ks = 0; ks < KSPLIT; ++ks) {
        h += part_h[(size_t)ks * n + i];
        s += part_s[(size_t)ks * n + i];
    }
    float r = fmaxf(h * __expf(fminf(s, 0.f)), 0.f);   // relu(h * delta)
    hout[i] = r;
    if (Hb) act_to_bf(r, Hb, Lb, i);
}

// ---- heads: y = h0@head0^T + h1@head1^T  (N=10, trivial reduction) --------
__global__ void __launch_bounds__(256)
head_gemm(const float* __restrict__ h0, const float* __restrict__ W0,
          const float* __restrict__ h1, const float* __restrict__ W1,
          float* __restrict__ y)
{
    const int b = blockIdx.x, t = threadIdx.x;
    float acc[NC];
#pragma unroll
    for (int c = 0; c < NC; ++c) acc[c] = 0.f;
    for (int o = t; o < O_DIM; o += 256) {
        float a0 = h0[(size_t)b * O_DIM + o];
        float a1 = h1[(size_t)b * O_DIM + o];
#pragma unroll
        for (int c = 0; c < NC; ++c)
            acc[c] += a0 * W0[(size_t)c * O_DIM + o]
                    + a1 * W1[(size_t)c * O_DIM + o];
    }
    __shared__ float red[256];
    for (int c = 0; c < NC; ++c) {
        red[t] = acc[c]; __syncthreads();
        for (int s = 128; s > 0; s >>= 1) {
            if (t < s) red[t] += red[t + s];
            __syncthreads();
        }
        if (t == 0) y[b * NC + c] = red[0];
        __syncthreads();
    }
}

extern "C" void kernel_launch(void* const* d_in, const int* in_sizes, int n_in,
                              void* d_out, int out_size, void* d_ws, size_t ws_size,
                              hipStream_t stream) {
    const float* x     = (const float*)d_in[0];
    const float* v0    = (const float*)d_in[1];
    const float* fc0   = (const float*)d_in[2];
    const float* head0 = (const float*)d_in[3];
    const float* v1    = (const float*)d_in[4];
    const float* fc1   = (const float*)d_in[5];
    const float* head1 = (const float*)d_in[6];
    float* y = (float*)d_out;

    char*  ws  = (char*)d_ws;
    size_t off = 0;
    auto alloc = [&](size_t bytes) {
        void* p = ws + off;
        off += (bytes + 255) & ~(size_t)255;
        return p;
    };
    const int nBH = BB * O_DIM;                                  // 65536
    float*          part_h = (float*)alloc(sizeof(float) * KSPLIT * nBH);
    float*          part_s = (float*)alloc(sizeof(float) * KSPLIT * nBH);
    unsigned short* Xb     = (unsigned short*)alloc(sizeof(short) * BB * K_DIM);
    unsigned short* Lb     = (unsigned short*)alloc(sizeof(short) * BB * K_DIM);
    float*          h0     = (float*)alloc(sizeof(float) * nBH);
    unsigned short* Hb     = (unsigned short*)alloc(sizeof(short) * nBH);
    unsigned short* L1b    = (unsigned short*)alloc(sizeof(short) * nBH);
    float*          h1     = (float*)alloc(sizeof(float) * nBH);

    const int nBI = BB * K_DIM;
    prep_input<<<(nBI + 255) / 256, 256, 0, stream>>>(x, Xb, Lb, nBI);

    // layer 0: 16 column blocks x KSPLIT K-slices, 4 waves/block
    harsanyi_gemm<<<dim3(O_DIM / 64, KSPLIT), 128, 0, stream>>>(
        Xb, Lb, v0, fc0, part_h, part_s);
    combine<<<(nBH + 255) / 256, 256, 0, stream>>>(
        part_h, part_s, h0, Hb, L1b, nBH);

    // layer 1
    harsanyi_gemm<<<dim3(O_DIM / 64, KSPLIT), 128, 0, stream>>>(
        Hb, L1b, v1, fc1, part_h, part_s);
    combine<<<(nBH + 255) / 256, 256, 0, stream>>>(
        part_h, part_s, h1, (unsigned short*)nullptr, (unsigned short*)nullptr, nBH);

    // heads -> y [64,10]
    head_gemm<<<BB, 256, 0, stream>>>(h0, head0, h1, head1, y);
}
